// QKVAttentionLegacy_66511863546392
// MI455X (gfx1250) — matrix operand order repaired and checked by reading, not compiled, not run on hardware
//
#include <hip/hip_runtime.h>
#include <hip/hip_bf16.h>
#include <hip/hip_fp16.h>

typedef __attribute__((ext_vector_type(16))) _Float16 v16h;
typedef __attribute__((ext_vector_type(8)))  _Float16 v8h;
typedef __attribute__((ext_vector_type(4)))  _Float16 v4h;
typedef __attribute__((ext_vector_type(8)))  float    v8f;

// LDS row strides (f16 units unless noted); padded, 16B-aligned strides
#define QS 72    // qS[i][dd]   128x64 (+8 pad)
#define KS 72    // kS[j][dd]   64x64  (+8 pad)
#define VS 72    // vS[dd][j]   80x64  (+8 pad); rows 64..79 = mask row + zeros
#define PS 72    // pS[row][j]  16x64 per wave (+8 pad)
#define OS 132   // oS[dd][i]   64x128 f32 (+4 pad)

__device__ __forceinline__ v16h make_frag(const _Float16* lo, const _Float16* hi) {
    v8h a = *(const v8h*)lo;   // ds_load_b128
    v8h b = *(const v8h*)hi;   // ds_load_b128
    v16h r;
#pragma unroll
    for (int i = 0; i < 8; ++i) { r[i] = a[i]; r[i + 8] = b[i]; }
    return r;
}

__device__ __forceinline__ v8f wmma_f16(v16h a, v16h b, v8f c) {
    return __builtin_amdgcn_wmma_f32_16x16x32_f16(false, a, false, b, (short)0, c,
                                                  false, false);
}

// DPP16 lane permute (VALU-only, no LDS)
template <int CTRL>
__device__ __forceinline__ float dppf(float x) {
    int xi = __builtin_bit_cast(int, x);
    int r  = __builtin_amdgcn_update_dpp(xi, xi, CTRL, 0xf, 0xf, true);
    return __builtin_bit_cast(float, r);
}

// butterfly reductions across the 16-lane group holding one C-fragment row
__device__ __forceinline__ float rowmax16(float v) {
    v = fmaxf(v, dppf<0xB1>(v));    // quad_perm xor1
    v = fmaxf(v, dppf<0x4E>(v));    // quad_perm xor2
    v = fmaxf(v, dppf<0x141>(v));   // row_half_mirror
    v = fmaxf(v, dppf<0x140>(v));   // row_mirror
    return v;
}
__device__ __forceinline__ float rowsum16(float v) {
    v += dppf<0xB1>(v);
    v += dppf<0x4E>(v);
    v += dppf<0x141>(v);
    v += dppf<0x140>(v);
    return v;
}

__global__ __launch_bounds__(256)
void qkv_attn_flash_wmma(const float* __restrict__ qkv,
                         const unsigned char* __restrict__ mask,
                         float* __restrict__ out) {
    constexpr int   N   = 2048;
    constexpr int   D   = 64;
    constexpr int   BM  = 128;  // query rows per block (8 waves x 16)
    constexpr int   BN  = 64;   // key cols per iteration
    // 1/sqrt(sqrt(64)) * sqrt(log2(e)) : softmax runs in the log2 domain
    constexpr float SCALE = 0.4246609001440095f;
    constexpr float NEG   = -3.0e38f;

    // qS stays live across the whole kernel; kS/vS/pS are dead after the loop
    // and their pool is reused for the f32 output-transpose buffer oS.
    constexpr int KS_B = 2 * BN * KS;        //  9216 B
    constexpr int VS_B = 2 * 80 * VS;        // 11520 B (64 data + mask + 15 zero rows)
    constexpr int PS_B = 2 * 8 * 16 * PS;    // 18432 B
    __shared__ __align__(16) _Float16 qS[BM * QS];                 // 18432 B
    __shared__ __align__(16) unsigned char poolS[KS_B + VS_B + PS_B]; // 39168 B
    __shared__ __align__(16) unsigned char mS[N];                  //  2048 B

    _Float16* kS = (_Float16*)poolS;
    _Float16* vS = (_Float16*)(poolS + KS_B);
    _Float16* pS = (_Float16*)(poolS + KS_B + VS_B);
    float*    oS = (float*)poolS;            // 64*132 f32 = 33792 B (aliases all)

    const int tid  = threadIdx.x;
    const int wave = tid >> 5;
    const int lane = tid & 31;
    const int lm   = lane & 15;   // column-in-tile (B/C) or row (A)
    const int hh   = lane >> 4;   // half-wave select

    const int bh    = blockIdx.x & 63;        // batch*8 + head
    const int iblk  = blockIdx.x >> 6;        // 0..15
    const int batch = bh >> 3, head = bh & 7;
    const int i0    = iblk * BM;

    const float* base = qkv + ((size_t)batch * 1536 + (size_t)head * 192) * N;
    const float* qp   = base;
    const float* kp   = base + (size_t)64 * N;
    const float* vp   = base + (size_t)128 * N;
    const unsigned char* mp = mask + (size_t)batch * N;

    // ---- stage Q (scaled f16, [i][dd]), mask row, zero vS rows 65..79 ----
    for (int idx = tid; idx < (BM * D) / 4; idx += 256) {
        int dd = idx >> 5, ii = (idx & 31) * 4;
        const float4 q4 = *(const float4*)(qp + (size_t)dd * N + i0 + ii);
        qS[(ii + 0) * QS + dd] = (_Float16)(q4.x * SCALE);
        qS[(ii + 1) * QS + dd] = (_Float16)(q4.y * SCALE);
        qS[(ii + 2) * QS + dd] = (_Float16)(q4.z * SCALE);
        qS[(ii + 3) * QS + dd] = (_Float16)(q4.w * SCALE);
    }
    for (int j = tid; j < N; j += 256) mS[j] = mp[j];
    for (int idx = tid; idx < 15 * BN; idx += 256) {      // one-time zero fill
        int r = idx >> 6, x = idx & 63;
        vS[(65 + r) * VS + x] = (_Float16)0.0f;
    }
    __syncthreads();

    // per-wave Q A-fragments (M=i row, K=dd), ISA 16-bit A layout
    const int qrow = wave * 16 + lm;
    const v16h aq0 = make_frag(&qS[qrow * QS +  0 + 8 * hh],
                               &qS[qrow * QS + 16 + 8 * hh]);
    const v16h aq1 = make_frag(&qS[qrow * QS + 32 + 8 * hh],
                               &qS[qrow * QS + 48 + 8 * hh]);

    // acc[0..3] = output tiles; acc[4] = softmax denominator (mask-ones column)
    v8f acc[5] = {v8f{}, v8f{}, v8f{}, v8f{}, v8f{}};
    float mrow[8];
#pragma unroll
    for (int v = 0; v < 8; ++v) mrow[v] = NEG;

    _Float16* pW = &pS[wave * 16 * PS];

    for (int jb = 0; jb < N; jb += BN) {
        __syncthreads();
        // ---- stage K (scaled) as kS[j][dd]; V (mask-zeroed) as vS[dd][j] ----
        for (int idx = tid; idx < (D * BN) / 4; idx += 256) {
            int dd = idx >> 4, j = (idx & 15) * 4;
            const float4 k4 = *(const float4*)(kp + (size_t)dd * N + jb + j);
            const float4 w4 = *(const float4*)(vp + (size_t)dd * N + jb + j);
            const uchar4 m4 = *(const uchar4*)(&mS[jb + j]);
            kS[(j + 0) * KS + dd] = (_Float16)(k4.x * SCALE);
            kS[(j + 1) * KS + dd] = (_Float16)(k4.y * SCALE);
            kS[(j + 2) * KS + dd] = (_Float16)(k4.z * SCALE);
            kS[(j + 3) * KS + dd] = (_Float16)(k4.w * SCALE);
            v4h vv = { (_Float16)(m4.x ? w4.x : 0.0f),
                       (_Float16)(m4.y ? w4.y : 0.0f),
                       (_Float16)(m4.z ? w4.z : 0.0f),
                       (_Float16)(m4.w ? w4.w : 0.0f) };
            *(v4h*)&vS[dd * VS + j] = vv;   // ds_store_b64
        }
        if (tid < 16) {   // mask indicator row (vS row 64) for the denominator tile
            const uchar4 m4 = *(const uchar4*)(&mS[jb + tid * 4]);
            v4h mm = { (_Float16)(m4.x ? 1.0f : 0.0f),
                       (_Float16)(m4.y ? 1.0f : 0.0f),
                       (_Float16)(m4.z ? 1.0f : 0.0f),
                       (_Float16)(m4.w ? 1.0f : 0.0f) };
            *(v4h*)&vS[64 * VS + tid * 4] = mm;
        }
        if (jb + BN < N) {  // global_prefetch_b8 of the next tiles
            const size_t off = (size_t)(tid >> 5) * 8 * N + 2 * (size_t)(tid & 31);
            __builtin_prefetch(kp + jb + BN + off, 0, 1);
            __builtin_prefetch(vp + jb + BN + off, 0, 1);
        }
        __syncthreads();

        // ---- S(log2 dom) = Qa*K, inline-0 C operand (no setup cost) ----
        v8f s[4];
#pragma unroll
        for (int c = 0; c < 4; ++c) {
            v16h bk0 = make_frag(&kS[(16 * c + lm) * KS +  0 + 8 * hh],
                                 &kS[(16 * c + lm) * KS + 16 + 8 * hh]);
            v16h bk1 = make_frag(&kS[(16 * c + lm) * KS + 32 + 8 * hh],
                                 &kS[(16 * c + lm) * KS + 48 + 8 * hh]);
            v8f z = {};
            z = wmma_f16(aq0, bk0, z);
            s[c] = wmma_f16(aq1, bk1, z);
        }

        // ---- online softmax: raw-score row max (shift cancels in normalize) ----
#pragma unroll
        for (int v = 0; v < 8; ++v) {
            float mx = rowmax16(fmaxf(fmaxf(s[0][v], s[1][v]),
                                      fmaxf(s[2][v], s[3][v])));
            float mn = fmaxf(mrow[v], mx);
            float al = __builtin_amdgcn_exp2f(mrow[v] - mn);   // bare v_exp_f32
            float p0 = __builtin_amdgcn_exp2f(s[0][v] - mn);
            float p1 = __builtin_amdgcn_exp2f(s[1][v] - mn);
            float p2 = __builtin_amdgcn_exp2f(s[2][v] - mn);
            float p3 = __builtin_amdgcn_exp2f(s[3][v] - mn);
            mrow[v] = mn;
#pragma unroll
            for (int t = 0; t < 5; ++t) acc[t][v] *= al;
            int prow = v + 8 * hh;
            pW[prow * PS +      lm] = (_Float16)p0;
            pW[prow * PS + 16 + lm] = (_Float16)p1;
            pW[prow * PS + 32 + lm] = (_Float16)p2;
            pW[prow * PS + 48 + lm] = (_Float16)p3;
        }
        asm volatile("" ::: "memory");  // keep LDS store->load order (in-order per wave)

        // ---- acc += P * V (t=0..3 output, t=4 denominator via mask row) ----
        v16h ap0 = make_frag(&pW[lm * PS +  0 + 8 * hh], &pW[lm * PS + 16 + 8 * hh]);
        v16h ap1 = make_frag(&pW[lm * PS + 32 + 8 * hh], &pW[lm * PS + 48 + 8 * hh]);
#pragma unroll
        for (int t = 0; t < 5; ++t) {
            v16h bv0 = make_frag(&vS[(16 * t + lm) * VS +  0 + 8 * hh],
                                 &vS[(16 * t + lm) * VS + 16 + 8 * hh]);
            v16h bv1 = make_frag(&vS[(16 * t + lm) * VS + 32 + 8 * hh],
                                 &vS[(16 * t + lm) * VS + 48 + 8 * hh]);
            acc[t] = wmma_f16(ap0, bv0, acc[t]);
            acc[t] = wmma_f16(ap1, bv1, acc[t]);
        }
    }

    __syncthreads();   // kS/vS/pS dead; pool becomes oS

    // ---- denominator broadcast, normalize (v_rcp), transpose via LDS ----
#pragma unroll
    for (int v = 0; v < 8; ++v) {
        float inv = __builtin_amdgcn_rcpf(rowsum16(acc[4][v]));
        int ii = wave * 16 + v + 8 * hh;
#pragma unroll
        for (int t = 0; t < 4; ++t)
            oS[(16 * t + lm) * OS + ii] = acc[t][v] * inv;
    }
    __syncthreads();
    float* op = out + ((size_t)batch * 512 + (size_t)head * 64) * N + i0;
    for (int idx = tid; idx < (D * BM) / 4; idx += 256) {
        int dd = idx >> 5, ii = (idx & 31) * 4;
        *(float4*)(op + (size_t)dd * N + ii) = *(const float4*)(&oS[dd * OS + ii]);
    }
}

extern "C" void kernel_launch(void* const* d_in, const int* in_sizes, int n_in,
                              void* d_out, int out_size, void* d_ws, size_t ws_size,
                              hipStream_t stream) {
    (void)in_sizes; (void)n_in; (void)out_size; (void)d_ws; (void)ws_size;
    const float*         qkv  = (const float*)d_in[0];
    const unsigned char* mask = (const unsigned char*)d_in[1];  // JAX bool -> 1 byte
    float*               out  = (float*)d_out;
    // 64 (batch*head) x 16 query-blocks of 128 rows
    dim3 grid(1024), block(256);
    qkv_attn_flash_wmma<<<grid, block, 0, stream>>>(qkv, mask, out);
}
